// TitansL2_57750130262093
// MI455X (gfx1250) — matrix operand order repaired and checked by reading, not compiled, not run on hardware
//
#include <hip/hip_runtime.h>
#include <hip/hip_bf16.h>

// Titans-L2 forward for MI455X (gfx1250, wave32).
// Sizes: B=1, T=2048, C=1024, H=16, D=64, CHUNK=128, NC=16.
// d_out = y (2048*1024 f32) ++ M_final (16*64*64 f32).
// Workspace usage: 11,534,336 floats = ~46.1 MB (Q,K,V,Y + Ach,Bch,M0).

#define TSEQ   2048
#define CEMB   1024
#define NHEAD  16
#define DHEAD  64
#define CSZ    128
#define NCH    16
#define Y_ELEMS (TSEQ * CEMB)

typedef __attribute__((ext_vector_type(16))) __bf16   v16bf;
typedef __attribute__((ext_vector_type(8)))  float    v8f;
typedef __attribute__((ext_vector_type(2)))  float    v2f;
typedef __attribute__((ext_vector_type(8)))  unsigned v8u;

__device__ inline unsigned short f2bf(float f) {
  unsigned u = __builtin_bit_cast(unsigned, f);
  unsigned r = u + 0x7FFFu + ((u >> 16) & 1u);   // round-to-nearest-even
  return (unsigned short)(r >> 16);
}
__device__ inline unsigned pack_bf16x2(float a, float b) {
  return (unsigned)f2bf(a) | ((unsigned)f2bf(b) << 16);
}
__device__ inline float sig_half(float x) {
  return 0.5f / (1.0f + __expf(-x));
}

// --- gfx1250 async copy: 16 bytes global -> LDS per active lane ------------
// ISA 15.18.3 op 98 GLOBAL_LOAD_ASYNC_TO_LDS_B128 (GV mode, saddr=off).
// VDST VGPR holds the LDS byte address (low 32 bits of the flat LDS address).
__device__ inline void async_ld16(unsigned lds_addr, const float* gaddr) {
  asm volatile("global_load_async_to_lds_b128 %0, %1, off"
               :: "v"(lds_addr), "v"(gaddr) : "memory");
}
__device__ inline void wait_async_le(int n) {   // ASYNCcnt <= n (n: 0..3 used)
  switch (n) {
    case 0: asm volatile("s_wait_asynccnt 0x0" ::: "memory"); break;
    case 2: asm volatile("s_wait_asynccnt 0x2" ::: "memory"); break;
    default: asm volatile("s_wait_asynccnt 0x3" ::: "memory"); break;
  }
}
__device__ inline unsigned lds_off(const void* p) {
  return (unsigned)(size_t)p;   // LDS flat addr: offset lives in bits [31:0]
}

// ---------------------------------------------------------------------------
// GEMM: Out[2048][1024] = X[2048][1024] * W[1024][1024]^T  (out[t,o]=sum_c x*W[o,c])
// bf16 WMMA 16x16x32, f32 accumulate. Block = 64x64 tile, 8 waves, 2 tiles/wave.
// ---------------------------------------------------------------------------
__global__ __launch_bounds__(256)
void titans_gemm_xwT(const float* __restrict__ X, const float* __restrict__ W,
                     float* __restrict__ Out) {
  __shared__ unsigned lA[64][16];  // X tile: [row][kpair] packed bf16x2
  __shared__ unsigned lB[64][16];  // W tile: [o]  [kpair] packed bf16x2

  const int tid  = threadIdx.x;
  const int lane = tid & 31;
  const int wave = tid >> 5;
  const int m0   = blockIdx.x * 64;
  const int n0   = blockIdx.y * 64;
  const int tm   = (wave & 3) * 16;   // tile row
  const int tn   = (wave >> 2) * 32;  // 2 col tiles at tn, tn+16
  const int lrow = lane & 15;
  const int kh   = lane >> 4;

  const int lr = tid >> 2;            // load row 0..63
  const int lc = (tid & 3) * 8;       // 8 consecutive floats

  v8f acc0 = {}; v8f acc1 = {};

  for (int k0 = 0; k0 < CEMB; k0 += 32) {
    const float4* xp = (const float4*)(X + (size_t)(m0 + lr) * CEMB + k0 + lc);
    const float4* wp = (const float4*)(W + (size_t)(n0 + lr) * CEMB + k0 + lc);
    float4 x0 = xp[0], x1 = xp[1];
    float4 w0 = wp[0], w1 = wp[1];
    if (k0 + 32 < CEMB) {             // global_prefetch_b8 next K tile
      __builtin_prefetch(X + (size_t)(m0 + lr) * CEMB + k0 + 32 + lc, 0, 1);
      __builtin_prefetch(W + (size_t)(n0 + lr) * CEMB + k0 + 32 + lc, 0, 1);
    }
    lA[lr][(lc >> 1) + 0] = pack_bf16x2(x0.x, x0.y);
    lA[lr][(lc >> 1) + 1] = pack_bf16x2(x0.z, x0.w);
    lA[lr][(lc >> 1) + 2] = pack_bf16x2(x1.x, x1.y);
    lA[lr][(lc >> 1) + 3] = pack_bf16x2(x1.z, x1.w);
    lB[lr][(lc >> 1) + 0] = pack_bf16x2(w0.x, w0.y);
    lB[lr][(lc >> 1) + 1] = pack_bf16x2(w0.z, w0.w);
    lB[lr][(lc >> 1) + 2] = pack_bf16x2(w1.x, w1.y);
    lB[lr][(lc >> 1) + 3] = pack_bf16x2(w1.z, w1.w);
    __syncthreads();

    // A frag (16x32 bf16): lane holds row tm+lrow; VGPR j -> K pair
    //   p = (j>>2)*8 + kh*4 + (j&3)    (ISA 7.12.2, 16-bit A 16x32)
    // B frag (32x16 bf16): lane holds col; pair p = (j>>2)*8 + kh*4 + (j&3)
    v8u au, bu0, bu1;
#pragma unroll
    for (int j = 0; j < 8; ++j) {
      int pa = ((j >> 2) << 3) + (kh << 2) + (j & 3);
      au[j]  = lA[tm + lrow][pa];
      int pb = ((j >> 2) << 3) + (kh << 2) + (j & 3);
      bu0[j] = lB[tn + lrow][pb];
      bu1[j] = lB[tn + 16 + lrow][pb];
    }
    v16bf a  = __builtin_bit_cast(v16bf, au);
    v16bf b0 = __builtin_bit_cast(v16bf, bu0);
    v16bf b1 = __builtin_bit_cast(v16bf, bu1);
    acc0 = __builtin_amdgcn_wmma_f32_16x16x32_bf16(false, a, false, b0,
                                                   (short)0, acc0, false, false);
    acc1 = __builtin_amdgcn_wmma_f32_16x16x32_bf16(false, a, false, b1,
                                                   (short)0, acc1, false, false);
    __syncthreads();
  }
  // C/D layout: VGPR r -> row r (lanes 0-15) / r+8 (lanes 16-31), col = lane&15
#pragma unroll
  for (int r = 0; r < 8; ++r) {
    int m = m0 + tm + r + kh * 8;
    Out[(size_t)m * CEMB + n0 + tn + lrow]      = acc0[r];
    Out[(size_t)m * CEMB + n0 + tn + 16 + lrow] = acc1[r];
  }
}

// ---------------------------------------------------------------------------
// k row-normalization: rows (t,h), 64 floats each; one wave per row.
// ---------------------------------------------------------------------------
__global__ __launch_bounds__(256)
void titans_knorm(float* __restrict__ K) {
  const int row  = blockIdx.x * 8 + (threadIdx.x >> 5);  // t*16+h
  const int lane = threadIdx.x & 31;
  float* p = K + (size_t)row * 64;
  float2 v = ((float2*)p)[lane];
  float ss = v.x * v.x + v.y * v.y;
#pragma unroll
  for (int m = 16; m >= 1; m >>= 1) ss += __shfl_xor(ss, m, 32);
  float scale = 1.0f / fmaxf(sqrtf(ss), 1e-12f);
  v.x *= scale; v.y *= scale;
  ((float2*)p)[lane] = v;
}

// ---------------------------------------------------------------------------
// op_step: per (head, chunk) sequential 128-step rank-1 recurrence.
// A,B (64x64 f32) live in registers: thread (d = tid>>2, q4 = tid&3) owns
// 16 columns e = q4*16..+15 of row d, for both A and B.
// k/v rows double-buffered in LDS, fed by async-to-LDS DMA (wave 0 issues,
// one b128 per lane over lanes 0..15 -> 64 floats per instruction).
// ---------------------------------------------------------------------------
__global__ __launch_bounds__(256)
void titans_opstep(const float* __restrict__ K, const float* __restrict__ V,
                   const float* __restrict__ alpha_raw,
                   const float* __restrict__ beta_raw,
                   float* __restrict__ Ach, float* __restrict__ Bch) {
  const int h  = blockIdx.x >> 4;
  const int n  = blockIdx.x & 15;
  const int tid = threadIdx.x;
  const int d  = tid >> 2;
  const int q4 = tid & 3;
  const float alpha = sig_half(alpha_raw[h]);
  const float beta  = sig_half(beta_raw[h]);

  float Areg[16], Breg[16];
#pragma unroll
  for (int i = 0; i < 16; ++i) {
    Areg[i] = ((q4 * 16 + i) == d) ? 1.0f : 0.0f;
    Breg[i] = 0.0f;
  }

  __shared__ float ks[2][64], vs[2][64];
  const float* Krow = K + (size_t)(n * CSZ) * CEMB + h * 64;
  const float* Vrow = V + (size_t)(n * CSZ) * CEMB + h * 64;

  if (tid < 16) {                      // prologue: stage t=0 into buffer 0
    async_ld16(lds_off(&ks[0][tid * 4]), Krow + tid * 4);
    async_ld16(lds_off(&vs[0][tid * 4]), Vrow + tid * 4);
  }

  for (int t = 0; t < CSZ; ++t) {
    const int buf = t & 1;
    __syncthreads();                   // everyone done reading buf^1 (iter t-1)
    if (t + 1 < CSZ && tid < 16) {     // overlap next row fetch with compute
      async_ld16(lds_off(&ks[buf ^ 1][tid * 4]), Krow + (size_t)(t + 1) * CEMB + tid * 4);
      async_ld16(lds_off(&vs[buf ^ 1][tid * 4]), Vrow + (size_t)(t + 1) * CEMB + tid * 4);
    }
    wait_async_le(t + 1 < CSZ ? 2 : 0);  // buf's 2 loads landed (in-order)
    __syncthreads();                   // publish LDS rows to all waves
    float ak = 0.f, bk = 0.f, kloc[16];
#pragma unroll
    for (int i = 0; i < 16; ++i) {
      kloc[i] = ks[buf][q4 * 16 + i];
      ak += Areg[i] * kloc[i];
      bk += Breg[i] * kloc[i];
    }
    ak += __shfl_xor(ak, 1, 32); ak += __shfl_xor(ak, 2, 32);
    bk += __shfl_xor(bk, 1, 32); bk += __shfl_xor(bk, 2, 32);
    const float ca = -alpha * ak;
    const float cb = -alpha * bk + beta * vs[buf][d];
#pragma unroll
    for (int i = 0; i < 16; ++i) {
      Areg[i] = fmaf(ca, kloc[i], Areg[i]);
      Breg[i] = fmaf(cb, kloc[i], Breg[i]);
    }
  }
  float* Ao = Ach + (size_t)(h * NCH + n) * 4096 + d * 64 + q4 * 16;
  float* Bo = Bch + (size_t)(h * NCH + n) * 4096 + d * 64 + q4 * 16;
#pragma unroll
  for (int i = 0; i < 16; ++i) { Ao[i] = Areg[i]; Bo[i] = Breg[i]; }
}

// ---------------------------------------------------------------------------
// chunk scan: per head, M <- M*A_n + B_n over 16 chunks, full f32 via
// v_wmma_f32_16x16x4_f32.  M (64x64) in LDS; 8 waves x 2 tiles of 16x16.
// Writes M0 (state at chunk start) and final M (to d_out tail).
// ---------------------------------------------------------------------------
__global__ __launch_bounds__(256)
void titans_chunkscan(const float* __restrict__ Ach, const float* __restrict__ Bch,
                      float* __restrict__ M0, float* __restrict__ Mfinal) {
  __shared__ float Ms[64][64];
  const int h    = blockIdx.x;
  const int tid  = threadIdx.x;
  const int lane = tid & 31;
  const int wave = tid >> 5;
  const int tm   = (wave & 3) * 16;
  const int tn   = (wave >> 2) * 32;
  const int lrow = lane & 15;
  const int kh   = lane >> 4;

  for (int i = tid; i < 4096; i += 256) Ms[i >> 6][i & 63] = 0.0f;
  __syncthreads();

  for (int n = 0; n < NCH; ++n) {
    const float* An = Ach + (size_t)(h * NCH + n) * 4096;
    const float* Bn = Bch + (size_t)(h * NCH + n) * 4096;
    float* M0p = M0 + (size_t)(h * NCH + n) * 4096;
    for (int i = tid; i < 4096; i += 256) M0p[i] = Ms[i >> 6][i & 63];

    v8f acc0, acc1;
#pragma unroll
    for (int r = 0; r < 8; ++r) {
      int dr = tm + r + kh * 8;
      acc0[r] = Bn[dr * 64 + tn + lrow];
      acc1[r] = Bn[dr * 64 + tn + 16 + lrow];
    }
#pragma unroll
    for (int kk = 0; kk < 16; ++kk) {
      // f32 A 16x4: VGPR0/1 = A[row][2*kh], A[row][2*kh+1]  (ISA 7.12.2)
      int e0 = kk * 4 + kh * 2;
      v2f va;  va[0]  = Ms[tm + lrow][e0];        va[1]  = Ms[tm + lrow][e0 + 1];
      v2f vb0; vb0[0] = An[e0 * 64 + tn + lrow];  vb0[1] = An[(e0 + 1) * 64 + tn + lrow];
      v2f vb1; vb1[0] = An[e0 * 64 + tn + 16 + lrow];
               vb1[1] = An[(e0 + 1) * 64 + tn + 16 + lrow];
      acc0 = __builtin_amdgcn_wmma_f32_16x16x4_f32(false, va, false, vb0,
                                                   (short)0, acc0, false, false);
      acc1 = __builtin_amdgcn_wmma_f32_16x16x4_f32(false, va, false, vb1,
                                                   (short)0, acc1, false, false);
    }
    __syncthreads();
#pragma unroll
    for (int r = 0; r < 8; ++r) {
      int dr = tm + r + kh * 8;
      Ms[dr][tn + lrow]      = acc0[r];
      Ms[dr][tn + 16 + lrow] = acc1[r];
    }
    __syncthreads();
  }
  for (int i = tid; i < 4096; i += 256) Mfinal[(size_t)h * 4096 + i] = Ms[i >> 6][i & 63];
}

// ---------------------------------------------------------------------------
// out_step: per (head, chunk): y_t = M q_t; M <- M(I-a k kT) + b v kT.
// Same register layout as op_step; q/k/v rows async-double-buffered;
// y written directly in (T, C) layout.
// ---------------------------------------------------------------------------
__global__ __launch_bounds__(256)
void titans_outstep(const float* __restrict__ Q, const float* __restrict__ K,
                    const float* __restrict__ V, const float* __restrict__ M0,
                    const float* __restrict__ alpha_raw,
                    const float* __restrict__ beta_raw,
                    float* __restrict__ Y) {
  const int h  = blockIdx.x >> 4;
  const int n  = blockIdx.x & 15;
  const int tid = threadIdx.x;
  const int d  = tid >> 2;
  const int q4 = tid & 3;
  const float alpha = sig_half(alpha_raw[h]);
  const float beta  = sig_half(beta_raw[h]);

  float Mreg[16];
  const float* m0p = M0 + (size_t)(h * NCH + n) * 4096 + d * 64 + q4 * 16;
#pragma unroll
  for (int i = 0; i < 16; ++i) Mreg[i] = m0p[i];

  __shared__ float qs[2][64], ks[2][64], vs[2][64];
  const size_t base = (size_t)(n * CSZ) * CEMB + h * 64;
  const float* Qrow = Q + base;
  const float* Krow = K + base;
  const float* Vrow = V + base;

  if (tid < 16) {
    async_ld16(lds_off(&qs[0][tid * 4]), Qrow + tid * 4);
    async_ld16(lds_off(&ks[0][tid * 4]), Krow + tid * 4);
    async_ld16(lds_off(&vs[0][tid * 4]), Vrow + tid * 4);
  }

  for (int t = 0; t < CSZ; ++t) {
    const int buf = t & 1;
    __syncthreads();
    if (t + 1 < CSZ && tid < 16) {
      size_t o = (size_t)(t + 1) * CEMB + tid * 4;
      async_ld16(lds_off(&qs[buf ^ 1][tid * 4]), Qrow + o);
      async_ld16(lds_off(&ks[buf ^ 1][tid * 4]), Krow + o);
      async_ld16(lds_off(&vs[buf ^ 1][tid * 4]), Vrow + o);
    }
    wait_async_le(t + 1 < CSZ ? 3 : 0);
    __syncthreads();
    float yp = 0.f, mk = 0.f, kloc[16];
#pragma unroll
    for (int i = 0; i < 16; ++i) {
      kloc[i] = ks[buf][q4 * 16 + i];
      yp += Mreg[i] * qs[buf][q4 * 16 + i];
      mk += Mreg[i] * kloc[i];
    }
    yp += __shfl_xor(yp, 1, 32); yp += __shfl_xor(yp, 2, 32);
    mk += __shfl_xor(mk, 1, 32); mk += __shfl_xor(mk, 2, 32);
    if (q4 == 0) Y[base + (size_t)t * CEMB + d] = yp;
    const float c = -alpha * mk + beta * vs[buf][d];
#pragma unroll
    for (int i = 0; i < 16; ++i) Mreg[i] = fmaf(c, kloc[i], Mreg[i]);
  }
}

// ---------------------------------------------------------------------------
extern "C" void kernel_launch(void* const* d_in, const int* in_sizes, int n_in,
                              void* d_out, int out_size, void* d_ws, size_t ws_size,
                              hipStream_t stream) {
  (void)in_sizes; (void)n_in; (void)out_size; (void)ws_size;
  const float* x  = (const float*)d_in[0];
  const float* Wq = (const float*)d_in[1];
  const float* Wk = (const float*)d_in[2];
  const float* Wv = (const float*)d_in[3];
  const float* Wp = (const float*)d_in[4];
  const float* ar = (const float*)d_in[5];
  const float* br = (const float*)d_in[6];
  float* out = (float*)d_out;

  float* ws  = (float*)d_ws;          // needs ~46.1 MB
  float* Q   = ws;
  float* K   = Q + Y_ELEMS;
  float* V   = K + Y_ELEMS;
  float* Y   = V + Y_ELEMS;
  float* Ach = Y + Y_ELEMS;           // 256 * 4096
  float* Bch = Ach + 256 * 4096;
  float* M0  = Bch + 256 * 4096;

  dim3 gg(TSEQ / 64, CEMB / 64), bb(256);
  titans_gemm_xwT<<<gg, bb, 0, stream>>>(x, Wq, Q);
  titans_gemm_xwT<<<gg, bb, 0, stream>>>(x, Wk, K);
  titans_gemm_xwT<<<gg, bb, 0, stream>>>(x, Wv, V);
  titans_knorm<<<(TSEQ * NHEAD) / 8, 256, 0, stream>>>(K);
  titans_opstep<<<NHEAD * NCH, 256, 0, stream>>>(K, V, ar, br, Ach, Bch);
  titans_chunkscan<<<NHEAD, 256, 0, stream>>>(Ach, Bch, M0, out + Y_ELEMS);
  titans_outstep<<<NHEAD * NCH, 256, 0, stream>>>(Q, K, V, M0, ar, br, Y);
  titans_gemm_xwT<<<gg, bb, 0, stream>>>(Y, Wp, out);
}
